// model_FNN_86663850099026
// MI455X (gfx1250) — matrix operand order repaired and checked by reading, not compiled, not run on hardware
//
#include <hip/hip_runtime.h>
#include <hip/hip_bf16.h>

// ---------------------------------------------------------------------------
// Problem constants (from the reference): u:(512,256,258) fp32,
// MLP 256->1024->1024->1024->1 with ReLU, then a length-512 scan over T.
// ---------------------------------------------------------------------------
#define T_DIM   512
#define B_DIM   256
#define F_DIM   258
#define IN_DIM  256
#define H_DIM   1024
#define NROWS   (T_DIM * B_DIM)        // 131072 rows through the MLP

#define M_TILE  64                      // rows per workgroup
#define STRIDE  1032                    // bf16 elems per LDS row (1024 + 8 pad)
#define BUF_BYTES (M_TILE * STRIDE * 2) // 132096 bytes per buffer
#define SMEM_BYTES (2 * BUF_BYTES)      // 264192 bytes dynamic LDS

typedef __bf16 bf16;
typedef __attribute__((ext_vector_type(16))) __bf16 bf16x16;
typedef __attribute__((ext_vector_type(8)))  float  floatx8;

union Frag {
    bf16x16 v;
    uint4   q[2];   // two 16-byte halves -> b128 loads
};

// ---------------------------------------------------------------------------
// Repack W (Ktot x N, fp32 row-major) into bf16 WMMA B-fragment order:
//   Wf[ ((ntile*KT + kt)*32 + lane)*16 + j ]
//     = W[ (kt*32 + 16*(lane>>4) + j) * N + ntile*16 + (lane&15) ]
// so a lane's 16 bf16 B-fragment elements are 32 contiguous bytes.
// (Contiguous k order, same for A: B 32x16, lanes 0-15 hold K=0..15, lanes 16-31 hold K=16..31.)
// ---------------------------------------------------------------------------
__global__ void prep_w_kernel(const float* __restrict__ W, bf16* __restrict__ Wf,
                              int Ktot, int N) {
    const int KT = Ktot >> 5;
    size_t o = (size_t)blockIdx.x * blockDim.x + threadIdx.x;
    size_t total = (size_t)Ktot * (size_t)N;
    if (o >= total) return;
    int j      = (int)(o & 15);
    int lane   = (int)((o >> 4) & 31);
    size_t rest = o >> 9;
    int kt     = (int)(rest % KT);
    int ntile  = (int)(rest / KT);
    int k = kt * 32 + 16 * (lane >> 4) + j;
    int n = ntile * 16 + (lane & 15);
    Wf[o] = (bf16)W[(size_t)k * N + n];
}

// ---------------------------------------------------------------------------
// One dense layer: out(64 x 1024) = relu( in(64 x 32*KT) @ W + bias ), bf16 in
// LDS, fp32 accumulate in WMMA. Each wave owns ntiles {wave, wave+8, ...};
// per ntile it holds 4 accumulators (row blocks 0..3) and loops K-tiles.
// A-fragment layout (contiguous k, as in Wf): lane m<16 = row m, elements
//   j=0..15 -> K = kt*32 + 16*half + j
//   (half = lane>=16)
// ---------------------------------------------------------------------------
__device__ __forceinline__ void gemm_layer(const bf16* __restrict__ in,
                                           bf16* __restrict__ out,
                                           const bf16* __restrict__ Wf,
                                           const float* __restrict__ bias,
                                           int KT, int lane, int wave) {
    const int lo   = lane & 15;
    const int half = lane >> 4;
    for (int nt = wave; nt < 64; nt += 8) {
        floatx8 acc0 = {}, acc1 = {}, acc2 = {}, acc3 = {};
        const bf16* wp = Wf + ((size_t)nt * KT) * 512 + lane * 16;
        for (int kt = 0; kt < KT; ++kt) {
            Frag b;
            b.q[0] = ((const uint4*)wp)[0];
            b.q[1] = ((const uint4*)wp)[1];
            wp += 512;
            __builtin_prefetch(wp, 0, 3);           // stream next K-tile of W

            const bf16* ap = in + kt * 32 + 16 * half;
            Frag a0, a1, a2, a3;
            a0.q[0] = *(const uint4*)(ap + (lo     ) * STRIDE);
            a0.q[1] = *(const uint4*)(ap + (lo     ) * STRIDE + 8);
            a1.q[0] = *(const uint4*)(ap + (16 + lo) * STRIDE);
            a1.q[1] = *(const uint4*)(ap + (16 + lo) * STRIDE + 8);
            a2.q[0] = *(const uint4*)(ap + (32 + lo) * STRIDE);
            a2.q[1] = *(const uint4*)(ap + (32 + lo) * STRIDE + 8);
            a3.q[0] = *(const uint4*)(ap + (48 + lo) * STRIDE);
            a3.q[1] = *(const uint4*)(ap + (48 + lo) * STRIDE + 8);

            acc0 = __builtin_amdgcn_wmma_f32_16x16x32_bf16(false, a0.v, false, b.v, (short)0, acc0, false, false);
            acc1 = __builtin_amdgcn_wmma_f32_16x16x32_bf16(false, a1.v, false, b.v, (short)0, acc1, false, false);
            acc2 = __builtin_amdgcn_wmma_f32_16x16x32_bf16(false, a2.v, false, b.v, (short)0, acc2, false, false);
            acc3 = __builtin_amdgcn_wmma_f32_16x16x32_bf16(false, a3.v, false, b.v, (short)0, acc3, false, false);
        }
        // Epilogue: C/D layout -> lane l: col = l&15, VGPR r: row = r + 8*(l>>4)
        const float bv  = bias[nt * 16 + lo];
        const int   col = nt * 16 + lo;
        const int   rb8 = 8 * half;
        #pragma unroll
        for (int r = 0; r < 8; ++r) {
            float v;
            v = acc0[r] + bv; out[( 0 + rb8 + r) * STRIDE + col] = (bf16)(v > 0.f ? v : 0.f);
            v = acc1[r] + bv; out[(16 + rb8 + r) * STRIDE + col] = (bf16)(v > 0.f ? v : 0.f);
            v = acc2[r] + bv; out[(32 + rb8 + r) * STRIDE + col] = (bf16)(v > 0.f ? v : 0.f);
            v = acc3[r] + bv; out[(48 + rb8 + r) * STRIDE + col] = (bf16)(v > 0.f ? v : 0.f);
        }
    }
}

// ---------------------------------------------------------------------------
// Fused MLP: 64 rows per workgroup, activations ping-pong in LDS (bf16).
// ---------------------------------------------------------------------------
__global__ __launch_bounds__(256, 1)
void mlp_kernel(const float* __restrict__ u,
                const bf16* __restrict__ Wf0, const bf16* __restrict__ Wf1,
                const bf16* __restrict__ Wf2,
                const float* __restrict__ b0, const float* __restrict__ b1,
                const float* __restrict__ b2,
                const float* __restrict__ W3, const float* __restrict__ b3,
                float* __restrict__ net) {
    extern __shared__ char smem[];
    bf16* bufA = (bf16*)smem;
    bf16* bufB = (bf16*)(smem + BUF_BYTES);

    const int tid  = threadIdx.x;
    const int lane = tid & 31;
    const int wave = tid >> 5;
    const size_t g0 = (size_t)blockIdx.x * M_TILE;

    // Load feats = u[:, :, 2:258] rows g0..g0+63 into bufA as bf16 (coalesced).
    for (int i = tid; i < M_TILE * IN_DIM; i += 256) {
        int r = i >> 8;          // row in tile
        int c = i & 255;         // feature column
        bufA[r * STRIDE + c] = (bf16)u[(g0 + r) * F_DIM + 2 + c];
    }
    __syncthreads();

    gemm_layer(bufA, bufB, Wf0, b0,  IN_DIM / 32, lane, wave);  // 256 -> 1024
    __syncthreads();
    gemm_layer(bufB, bufA, Wf1, b1,  H_DIM / 32, lane, wave);   // 1024 -> 1024
    __syncthreads();
    gemm_layer(bufA, bufB, Wf2, b2,  H_DIM / 32, lane, wave);   // 1024 -> 1024
    __syncthreads();

    // Final layer: net = h @ W3 + b3 (1024 -> 1). 4 threads per row.
    const int row = tid & 63;
    const int seg = tid >> 6;
    float p = 0.f;
    const bf16* hrow = bufB + row * STRIDE + seg * 256;
    #pragma unroll 4
    for (int c = 0; c < 256; ++c)
        p += (float)hrow[c] * W3[seg * 256 + c];

    float* red = (float*)bufA;   // bufA is free now
    red[tid] = p;
    __syncthreads();
    if (tid < 64) {
        float tot = red[tid] + red[tid + 64] + red[tid + 128] + red[tid + 192] + b3[0];
        net[g0 + tid] = tot;
    }
}

// ---------------------------------------------------------------------------
// Scan: y_t = a*y_{t-1} + b*c1_t + net_t, y_{-1} = u[0,:,0].
// One thread per batch column; 512 sequential steps (tiny).
// ---------------------------------------------------------------------------
__global__ void scan_kernel(const float* __restrict__ u,
                            const float* __restrict__ pa, const float* __restrict__ pb,
                            const float* __restrict__ net, float* __restrict__ out) {
    const int b = threadIdx.x;
    const float a  = *pa;
    const float bc = *pb;
    float y = u[(size_t)b * F_DIM + 0];   // u[0, b, 0]
    out[b] = y;
    for (int t = 0; t < T_DIM; ++t) {
        float c1 = u[((size_t)t * B_DIM + b) * F_DIM + 1];
        y = a * y + bc * c1 + net[t * B_DIM + b];
        out[(t + 1) * B_DIM + b] = y;
    }
}

// ---------------------------------------------------------------------------
// Workspace layout (bytes):
//   Wf0 bf16: 256*1024*2   =  524288   @ 0
//   Wf1 bf16: 1024*1024*2  = 2097152   @ 524288
//   Wf2 bf16: 1024*1024*2  = 2097152   @ 2621440
//   net fp32: 131072*4     =  524288   @ 4718592     (total ~5.0 MB)
// ---------------------------------------------------------------------------
extern "C" void kernel_launch(void* const* d_in, const int* in_sizes, int n_in,
                              void* d_out, int out_size, void* d_ws, size_t ws_size,
                              hipStream_t stream) {
    const float* u  = (const float*)d_in[0];
    const float* pa = (const float*)d_in[1];
    const float* pb = (const float*)d_in[2];
    const float* W0 = (const float*)d_in[3];
    const float* b0 = (const float*)d_in[4];
    const float* W1 = (const float*)d_in[5];
    const float* b1 = (const float*)d_in[6];
    const float* W2 = (const float*)d_in[7];
    const float* b2 = (const float*)d_in[8];
    const float* W3 = (const float*)d_in[9];
    const float* b3 = (const float*)d_in[10];
    float* out = (float*)d_out;

    char* ws = (char*)d_ws;
    bf16*  Wf0 = (bf16*)(ws + 0);
    bf16*  Wf1 = (bf16*)(ws + 524288);
    bf16*  Wf2 = (bf16*)(ws + 2621440);
    float* net = (float*)(ws + 4718592);

    // 1) repack weights to bf16 fragment layout
    prep_w_kernel<<<(IN_DIM * H_DIM) / 256, 256, 0, stream>>>(W0, Wf0, IN_DIM, H_DIM);
    prep_w_kernel<<<(H_DIM * H_DIM) / 256, 256, 0, stream>>>(W1, Wf1, H_DIM, H_DIM);
    prep_w_kernel<<<(H_DIM * H_DIM) / 256, 256, 0, stream>>>(W2, Wf2, H_DIM, H_DIM);

    // 2) fused MLP over all 131072 rows
    mlp_kernel<<<NROWS / M_TILE, 256, SMEM_BYTES, stream>>>(
        u, Wf0, Wf1, Wf2, b0, b1, b2, W3, b3, net);

    // 3) sequential scan (tiny)
    scan_kernel<<<1, B_DIM, 0, stream>>>(u, pa, pb, net, out);
}
